// Leray_Schauder_25683904430313
// MI455X (gfx1250) — matrix operand order repaired and checked by reading, not compiled, not run on hardware
//
#include <hip/hip_runtime.h>
#include <math.h>

#define DIM    8
#define NPTS   1024
#define NBASIS 8
#define HH     512
#define CH     16
#define BB     8

typedef float        v8f   __attribute__((ext_vector_type(8)));
typedef __bf16       v16bf __attribute__((ext_vector_type(16)));
typedef int          i4v   __attribute__((ext_vector_type(4)));
typedef int          i8vv  __attribute__((ext_vector_type(8)));
typedef unsigned int u4v   __attribute__((ext_vector_type(4)));

union frag32 { i4v q[2]; v16bf v; };

static __device__ __forceinline__ unsigned short f32_to_bf16(float f) {
    unsigned int u = __builtin_bit_cast(unsigned int, f);
    u += 0x7FFFu + ((u >> 16) & 1u);          // round-to-nearest-even
    return (unsigned short)(u >> 16);
}

static __device__ __forceinline__ float elu(float v) {
    return v > 0.0f ? v : (__expf(v) - 1.0f);
}

// Load a 16-element bf16 WMMA fragment from two 16-byte chunks.
static __device__ __forceinline__ v16bf ld_frag(const unsigned short* p0,
                                                const unsigned short* p1) {
    frag32 f;
    f.q[0] = *reinterpret_cast<const i4v*>(p0);
    f.q[1] = *reinterpret_cast<const i4v*>(p1);
    return f.v;
}

// ---------------------------------------------------------------------------
// prep 1: a0 = elu(s @ Wfirst + bfirst)  -> bf16 [NPTS][HH]
// ---------------------------------------------------------------------------
__global__ __launch_bounds__(256) void prep_a0(const float* __restrict__ s,
                                               const float* __restrict__ Wfirst,
                                               const float* __restrict__ bfirst,
                                               unsigned short* __restrict__ a0bf) {
    int idx = blockIdx.x * 256 + threadIdx.x;      // < NPTS*HH
    int m = idx >> 9, h = idx & (HH - 1);
    float acc = bfirst[h];
#pragma unroll
    for (int d = 0; d < DIM; ++d) acc += s[m * DIM + d] * Wfirst[d * HH + h];
    a0bf[idx] = f32_to_bf16(elu(acc));
}

// ---------------------------------------------------------------------------
// prep 2: WbT[i][n][h] = bf16(Wb[i][h][n])   (LDS-tiled transpose)
// ---------------------------------------------------------------------------
__global__ __launch_bounds__(256) void prep_wbT(const float* __restrict__ Wb,
                                                unsigned short* __restrict__ WbT) {
    __shared__ unsigned short tile[32][33];
    const int i  = blockIdx.z;
    const int bx = blockIdx.x * 32, by = blockIdx.y * 32;
    const int tx = threadIdx.x, ty = threadIdx.y;           // 32 x 8
    const float* src = Wb + (size_t)i * HH * HH;
    unsigned short* dst = WbT + (size_t)i * HH * HH;
#pragma unroll
    for (int j = 0; j < 4; ++j) {
        int h = by + ty + j * 8, n = bx + tx;
        tile[ty + j * 8][tx] = f32_to_bf16(src[(size_t)h * HH + n]);
    }
    __syncthreads();
#pragma unroll
    for (int j = 0; j < 4; ++j) {
        int n = bx + ty + j * 8, h = by + tx;
        dst[(size_t)n * HH + h] = tile[tx][ty + j * 8];
    }
}

// ---------------------------------------------------------------------------
// prep 3: fmod = s@Wf + bf - blast ; WlastT bf16 ; a0x ; zero accum
// ---------------------------------------------------------------------------
__global__ __launch_bounds__(256) void prep_misc(const float* __restrict__ s,
                                                 const float* __restrict__ x,
                                                 const float* __restrict__ Wfirst,
                                                 const float* __restrict__ bfirst,
                                                 const float* __restrict__ Wlast,
                                                 const float* __restrict__ blast,
                                                 const float* __restrict__ Wf,
                                                 const float* __restrict__ bfv,
                                                 unsigned short* __restrict__ WlastT,
                                                 float* __restrict__ fmod,
                                                 float* __restrict__ a0x,
                                                 float* __restrict__ accum) {
    int gid = blockIdx.x * 256 + threadIdx.x;
    if (gid < NPTS * CH) {
        int m = gid >> 4, c = gid & (CH - 1);
        float acc = bfv[c] - blast[c];
#pragma unroll
        for (int d = 0; d < DIM; ++d) acc += s[m * DIM + d] * Wf[d * CH + c];
        fmod[gid] = acc;
    } else if (gid < NPTS * CH + CH * HH) {
        int j = gid - NPTS * CH;
        int c = j >> 9, h = j & (HH - 1);
        WlastT[j] = f32_to_bf16(Wlast[h * CH + c]);
    } else if (gid < NPTS * CH + CH * HH + HH) {
        int h = gid - (NPTS * CH + CH * HH);
        float acc = bfirst[h];
#pragma unroll
        for (int d = 0; d < DIM; ++d) acc += x[d] * Wfirst[d * HH + h];
        a0x[h] = elu(acc);
    } else if (gid < NPTS * CH + CH * HH + HH + NBASIS * CH) {
        accum[gid - (NPTS * CH + CH * HH + HH)] = 0.0f;
    }
}

// ---------------------------------------------------------------------------
// main: per (m-tile, basis):
//   TDM-stage a0 tile -> LDS, a1 = elu(a0@Wb+bb) via pipelined bf16 WMMA,
//   z = a1@Wlast via bf16 WMMA (LDS staging), accumulate per-channel diff^2.
// ---------------------------------------------------------------------------
__global__ __launch_bounds__(256) void leray_main(const unsigned short* __restrict__ a0bf,
                                                  const unsigned short* __restrict__ WbT,
                                                  const unsigned short* __restrict__ WlastT,
                                                  const float* __restrict__ fmod,
                                                  const float* __restrict__ bb,
                                                  float* __restrict__ accum) {
    __shared__ alignas(16) unsigned short as0[16 * HH];   // 16 KB a0 tile (TDM dest)
    __shared__ alignas(16) unsigned short a1s[16 * HH];   // 16 KB elu(a1) tile
    __shared__ float zred[16 * CH];                       // 1 KB partial z

    const int tid  = threadIdx.x;
    const int wave = tid >> 5;
    const int lane = tid & 31;
    const int kh   = lane >> 4;       // K-half select within fragment
    const int l15  = lane & 15;
    const int mtile = blockIdx.x;
    const int basis = blockIdx.y;

    zred[tid] = 0.0f;                 // 256 == 16*CH

    // ---- stage the contiguous 16 KB a0 m-tile into LDS via the TDM ----
    const unsigned short* Asrc = a0bf + (size_t)mtile * 16 * HH;
#if __has_builtin(__builtin_amdgcn_tensor_load_to_lds)
    if (wave == 0) {
        unsigned lds_off = (unsigned)(size_t)(void*)as0;
        unsigned long long ga = (unsigned long long)(size_t)(const void*)Asrc;
        // D# group 0: count=1, lds_addr, global_addr, type=2 ("image")
        u4v g0 = (u4v)0u;
        g0[0] = 1u;
        g0[1] = lds_off;
        g0[2] = (unsigned)ga;
        g0[3] = (unsigned)((ga >> 32) & 0x1FFFFFFu) | (2u << 30);
        // D# group 1: data_size=8B, 1-D tile of 2048 elements (= 16 KB)
        i8vv g1 = (i8vv)0;
        g1[0] = 3 << 16;                           // data_size = 8 bytes
        g1[1] = (int)((2048u & 0xFFFFu) << 16);    // tensor_dim0 = 2048
        g1[2] = 1 << 16;                           // tensor_dim1 = 1
        g1[3] = (int)((2048u & 0xFFFFu) << 16);    // tile_dim0 = 2048
        g1[4] = 1;                                 // tile_dim1 = 1
        g1[5] = 2048;                              // tensor_dim0_stride = 2048
        i4v gz = (i4v)0;
#if defined(__clang_major__) && (__clang_major__ >= 23)
        i8vv g4 = (i8vv)0;
        __builtin_amdgcn_tensor_load_to_lds(g0, g1, gz, gz, g4, 0);
#else
        __builtin_amdgcn_tensor_load_to_lds(g0, g1, gz, gz, 0);
#endif
        __builtin_amdgcn_s_wait_tensorcnt(0);
    }
#else
    {   // fallback: cooperative vectorized copy
        const i4v* gsrc = reinterpret_cast<const i4v*>(Asrc);
        i4v* ldst = reinterpret_cast<i4v*>(as0);
#pragma unroll
        for (int j = 0; j < 4; ++j) ldst[tid + j * 256] = gsrc[tid + j * 256];
    }
#endif
    __syncthreads();

    // ---------------- GEMM1: 16 rows x 512 cols, K = 512 (pipelined) --------
    v8f acc[4] = {};
    const unsigned short* Als = as0 + l15 * HH;           // A rows from LDS
    const int aLo = kh * 8, aHi = 16 + kh * 8;            // A-layout K offsets
    const unsigned short* Bbase = WbT + (size_t)basis * HH * HH;
    const int n0 = wave * 64;                              // 4 N-tiles / wave
    const unsigned short* Brow[4];
#pragma unroll
    for (int t = 0; t < 4; ++t)
        Brow[t] = Bbase + (size_t)(n0 + t * 16 + l15) * HH + kh * 8;

    // prologue: fragments for kc = 0
    v16bf aF = ld_frag(Als + aLo, Als + aHi);
    v16bf bF[4];
#pragma unroll
    for (int t = 0; t < 4; ++t) bF[t] = ld_frag(Brow[t], Brow[t] + 16);

#pragma unroll 5
    for (int kc = 0; kc < HH - 32; kc += 32) {
        // issue next fragment set before consuming the current one
        v16bf aN = ld_frag(Als + kc + 32 + aLo, Als + kc + 32 + aHi);
        __builtin_prefetch(Brow[0] + kc + 64, 0, 1);
        v16bf bN[4];
#pragma unroll
        for (int t = 0; t < 4; ++t)
            bN[t] = ld_frag(Brow[t] + kc + 32, Brow[t] + kc + 48);
#pragma unroll
        for (int t = 0; t < 4; ++t)
            acc[t] = __builtin_amdgcn_wmma_f32_16x16x32_bf16(
                false, aF, false, bF[t], (short)0, acc[t], false, false);
        aF = aN;
#pragma unroll
        for (int t = 0; t < 4; ++t) bF[t] = bN[t];
    }
#pragma unroll
    for (int t = 0; t < 4; ++t)   // epilogue: last K-chunk
        acc[t] = __builtin_amdgcn_wmma_f32_16x16x32_bf16(
            false, aF, false, bF[t], (short)0, acc[t], false, false);

    // bias + elu, stash tile to LDS as bf16 (C layout: reg e -> M = e + kh*8)
    const float* bbp = bb + basis * HH;
#pragma unroll
    for (int t = 0; t < 4; ++t) {
        int n = n0 + t * 16 + l15;
        float bias = bbp[n];
#pragma unroll
        for (int e = 0; e < 8; ++e) {
            int m = e + kh * 8;
            a1s[m * HH + n] = f32_to_bf16(elu(acc[t][e] + bias));
        }
    }
    __syncthreads();

    // ---------------- GEMM2: z[16x16] = a1 @ Wlast, K split across waves ----
    v8f zacc = {};
    const int kb = wave * 64;
    const unsigned short* Wrow = WlastT + (size_t)l15 * HH;
#pragma unroll
    for (int kk = 0; kk < 2; ++kk) {
        int kc = kb + kk * 32;
        const unsigned short* ap = &a1s[l15 * HH + kc];
        v16bf af  = ld_frag(ap + aLo, ap + aHi);
        v16bf bfr = ld_frag(Wrow + kc + kh * 8, Wrow + kc + kh * 8 + 16);
        zacc = __builtin_amdgcn_wmma_f32_16x16x32_bf16(
            false, af, false, bfr, (short)0, zacc, false, false);
    }
#pragma unroll
    for (int e = 0; e < 8; ++e) {
        int m = e + kh * 8;
        atomicAdd(&zred[m * CH + l15], zacc[e]);          // ds_add_f32
    }
    __syncthreads();

    // ---------------- per-channel sum of (fmod - z)^2 ----------------
    {
        int m = tid >> 4, c = tid & (CH - 1);
        float d = fmod[(size_t)(mtile * 16 + m) * CH + c] - zred[tid];
        atomicAdd(&accum[basis * CH + c], d * d);          // global_atomic_add_f32
    }
}

// ---------------------------------------------------------------------------
// finalize: exact f32 x-path, mu gating, normalized sum, replicate over batch
// ---------------------------------------------------------------------------
__global__ __launch_bounds__(256) void finalize(const float* __restrict__ Wb,
                                                const float* __restrict__ bb,
                                                const float* __restrict__ Wlast,
                                                const float* __restrict__ blast,
                                                const float* __restrict__ a0x,
                                                const float* __restrict__ accum,
                                                float* __restrict__ out) {
    __shared__ float a1x[NBASIS][HH];   // 16 KB
    __shared__ float zx[NBASIS][CH];
    __shared__ float mu[NBASIS];
    const int tid = threadIdx.x;

    for (int idx = tid; idx < NBASIS * HH; idx += 256) {
        int i = idx >> 9, k = idx & (HH - 1);
        float acc = bb[i * HH + k];
        const float* wp = Wb + (size_t)i * HH * HH + k;
        for (int h = 0; h < HH; ++h) acc += a0x[h] * wp[(size_t)h * HH];
        a1x[i][k] = elu(acc);
    }
    __syncthreads();

    if (tid < NBASIS * CH) {
        int i = tid >> 4, c = tid & (CH - 1);
        float z = blast[c];
        for (int k = 0; k < HH; ++k) z += a1x[i][k] * Wlast[k * CH + c];
        zx[i][c] = z;
    } else if (tid < NBASIS * CH + NBASIS) {
        int i = tid - NBASIS * CH;
        float S = 0.0f;
        for (int c = 0; c < CH; ++c) S += accum[i * CH + c];
        // integral_c = VOLUME/NPTS * S_c = S_c/4 ; norm_b = sqrt(sum_c integral_c)
        float nb = 0.5f * sqrtf(S);
        mu[i] = (nb <= 1000.0f) ? nb : 0.0f;
    }
    __syncthreads();

    if (tid < CH) {
        float num = 0.0f, den = 1e-7f;
        for (int i = 0; i < NBASIS; ++i) { num += mu[i] * zx[i][tid]; den += mu[i]; }
        float val = num / den;
        for (int b = 0; b < BB; ++b) out[b * CH + tid] = val;   // batch-replicated
    }
}

// ---------------------------------------------------------------------------
extern "C" void kernel_launch(void* const* d_in, const int* in_sizes, int n_in,
                              void* d_out, int out_size, void* d_ws, size_t ws_size,
                              hipStream_t stream) {
    (void)in_sizes; (void)n_in; (void)out_size; (void)ws_size;
    const float* s      = (const float*)d_in[0];
    const float* x      = (const float*)d_in[1];
    const float* Wfirst = (const float*)d_in[2];
    const float* bfirst = (const float*)d_in[3];
    const float* Wb     = (const float*)d_in[4];
    const float* bb     = (const float*)d_in[5];
    const float* Wlast  = (const float*)d_in[6];
    const float* blast  = (const float*)d_in[7];
    const float* Wf     = (const float*)d_in[8];
    const float* bfv    = (const float*)d_in[9];
    float* out = (float*)d_out;

    char* ws = (char*)d_ws;
    unsigned short* a0bf   = (unsigned short*)(ws);             // 1,048,576 B
    unsigned short* WbT    = (unsigned short*)(ws + 1048576);   // 4,194,304 B
    unsigned short* WlastT = (unsigned short*)(ws + 5242880);   //    16,384 B
    float*          fmod   = (float*)(ws + 5259264);            //    65,536 B
    float*          a0x    = (float*)(ws + 5324800);            //     2,048 B
    float*          accum  = (float*)(ws + 5326848);            //       512 B

    prep_a0<<<(NPTS * HH) / 256, 256, 0, stream>>>(s, Wfirst, bfirst, a0bf);
    prep_wbT<<<dim3(HH / 32, HH / 32, NBASIS), dim3(32, 8), 0, stream>>>(Wb, WbT);
    prep_misc<<<(NPTS * CH + CH * HH + HH + NBASIS * CH + 255) / 256, 256, 0, stream>>>(
        s, x, Wfirst, bfirst, Wlast, blast, Wf, bfv, WlastT, fmod, a0x, accum);
    leray_main<<<dim3(NPTS / 16, NBASIS), 256, 0, stream>>>(a0bf, WbT, WlastT, fmod, bb, accum);
    finalize<<<1, 256, 0, stream>>>(Wb, bb, Wlast, blast, a0x, accum, out);
}
